// WindowedAttention_30142080483993
// MI455X (gfx1250) — matrix operand order repaired and checked by reading, not compiled, not run on hardware
//
#include <hip/hip_runtime.h>

typedef __attribute__((ext_vector_type(16))) __bf16 v16bf;
typedef __attribute__((ext_vector_type(8)))  __bf16 v8bf;
typedef __attribute__((ext_vector_type(8)))  float  v8f;

#define D_MODEL 128
#define SEQ_L   32768
#define WIN     64
#define NWIN    512        // SEQ_L / WIN
#define NHEAD   4
#define DH      32
#define T_PITCH 136        // bf16 elems per row (128 + 8 pad)
#define VT_PITCH 72        // bf16 elems per Vt row (64 + 8 pad)
#define S_PITCH 68         // f32 elems per score row (64 + 4 pad)

// LDS offsets (bytes)
#define OFF_Q  0
#define OFF_K  17408                 // 64*136*2
#define OFF_VT 34816                 // + 64*136*2
#define OFF_S  53248                 // + 128*72*2
#define LDS_BYTES 122880             // + 256*68*4

__device__ __forceinline__ v8f wmma_bf16(v16bf a, v16bf b, v8f c) {
    return __builtin_amdgcn_wmma_f32_16x16x32_bf16(false, a, false, b, (short)0, c,
                                                   false, false);
}

// A fragment (16x32 bf16), row-major source with pitch (bf16 elems).
// lane l: m = l&15, half g = l>>4; elems 0..15 at k0+g*16 (contiguous).
__device__ __forceinline__ v16bf loadA(const __bf16* base, int pitch, int row0,
                                       int k0, int lane) {
    int g = lane >> 4, m = lane & 15;
    const __bf16* p = base + (size_t)(row0 + m) * pitch + k0 + g * 16;
    union { v16bf v; v8bf h[2]; } u;
    u.h[0] = *(const v8bf*)p;
    u.h[1] = *(const v8bf*)(p + 8);
    return u.v;
}

// B fragment: 16 contiguous bf16 per lane starting at p (caller computes lane base).
__device__ __forceinline__ v16bf loadB16(const __bf16* p) {
    union { v16bf v; v8bf h[2]; } u;
    u.h[0] = *(const v8bf*)p;
    u.h[1] = *(const v8bf*)(p + 8);
    return u.v;
}

// D (16x16 f32) store to f32 LDS: lane n = l&15, m = (l>>4)*8 + j
__device__ __forceinline__ void storeD_f32(float* base, int pitch, int row0, int col0,
                                           v8f c, int lane) {
    int g = lane >> 4, n = lane & 15;
#pragma unroll
    for (int j = 0; j < 8; ++j)
        base[(size_t)(row0 + g * 8 + j) * pitch + col0 + n] = c[j];
}

__device__ __forceinline__ void storeD_bf16(__bf16* base, int pitch, int row0, int col0,
                                            v8f c, int lane, float scale) {
    int g = lane >> 4, n = lane & 15;
#pragma unroll
    for (int j = 0; j < 8; ++j)
        base[(size_t)(row0 + g * 8 + j) * pitch + col0 + n] = (__bf16)(c[j] * scale);
}

// store D transposed into Vt[col][token] (8 contiguous bf16 per lane -> one b128)
__device__ __forceinline__ void storeD_vt(__bf16* vt, int row0, int col0, v8f c, int lane) {
    int g = lane >> 4, n = lane & 15;
    union { v8bf v; __bf16 e[8]; } u;
#pragma unroll
    for (int j = 0; j < 8; ++j) u.e[j] = (__bf16)c[j];
    *(v8bf*)(vt + (size_t)(col0 + n) * VT_PITCH + row0 + g * 8) = u.v;
}

// store Y[d][w] f32 (transposed D): 8 contiguous f32 per lane -> two b128
__device__ __forceinline__ void storeD_y(float* y, int row0, int col0, v8f c, int lane) {
    int g = lane >> 4, n = lane & 15;
    float* p = y + (size_t)(col0 + n) * S_PITCH + row0 + g * 8;
    float4 a = {c[0], c[1], c[2], c[3]};
    float4 b = {c[4], c[5], c[6], c[7]};
    *(float4*)p = a;
    *(float4*)(p + 4) = b;
}

// Pack W[128][128] f32 -> bf16 B-fragment layout:
// frag(kt 0..3, nt 0..7): 512 bf16 = lane*16 + i, n = nt*16 + (lane&15),
// k = kt*32 + (lane>>4)*16 + i
__global__ void wa_pack_weights(const float* __restrict__ Wq, const float* __restrict__ Wk,
                                const float* __restrict__ Wv, const float* __restrict__ Wo,
                                __bf16* __restrict__ wpack) {
    int idx = blockIdx.x * 256 + threadIdx.x;   // 65536 total
    int wi   = idx >> 14;
    int rem  = idx & 16383;
    int frag = rem >> 9;
    int lane = (rem >> 4) & 31;
    int i    = rem & 15;
    int kt = frag >> 3, nt = frag & 7;
    int n = nt * 16 + (lane & 15);
    int k = kt * 32 + (lane >> 4) * 16 + i;
    const float* W = (wi == 0) ? Wq : (wi == 1) ? Wk : (wi == 2) ? Wv : Wo;
    wpack[idx] = (__bf16)W[k * D_MODEL + n];
}

__global__ __launch_bounds__(256) void wa_window_kernel(const float* __restrict__ x,
                                                        const __bf16* __restrict__ wpack,
                                                        float* __restrict__ out) {
    extern __shared__ char smem[];
    __bf16* Qs  = (__bf16*)(smem + OFF_Q);
    __bf16* Ks  = (__bf16*)(smem + OFF_K);
    __bf16* Vts = (__bf16*)(smem + OFF_VT);
    float*  Ss  = (float*)(smem + OFF_S);
    __bf16* Ts  = (__bf16*)(smem + OFF_S);   // phase-1 overlay
    __bf16* Ps  = (__bf16*)(smem + OFF_S);   // softmax writes bf16 in place
    __bf16* Os  = (__bf16*)(smem + OFF_Q);   // phase-5 overlay (Q dead)
    float*  Ys  = (float*)(smem + OFF_S);    // phase-6 overlay (P dead)

    const int tid  = threadIdx.x;
    const int lane = tid & 31;
    const int wave = tid >> 5;
    const int win  = blockIdx.x;
    const int b    = win >> 9;        // / NWIN
    const int wn   = win & (NWIN - 1);
    const float scale = 0.17677669529663689f;   // 1/sqrt(32)

    // ---- Phase 1: load token tile T[w][d] = x[b, d, wn*64 + w], convert to bf16
    {
        int d = tid >> 1, half = tid & 1;
        const float* src = x + ((size_t)b * D_MODEL + d) * SEQ_L + wn * WIN + half * 32;
#pragma unroll
        for (int c = 0; c < 8; ++c) {
            float4 v = *(const float4*)(src + c * 4);
            int w = half * 32 + c * 4;
            Ts[(size_t)(w + 0) * T_PITCH + d] = (__bf16)v.x;
            Ts[(size_t)(w + 1) * T_PITCH + d] = (__bf16)v.y;
            Ts[(size_t)(w + 2) * T_PITCH + d] = (__bf16)v.z;
            Ts[(size_t)(w + 3) * T_PITCH + d] = (__bf16)v.w;
        }
    }
    __syncthreads();

    // ---- Phase 2: Q = T*Wq (scaled), K = T*Wk, Vt = (T*Wv)^T
    // Each wave owns column tile nt = wave. B fragments for all three matrices are
    // held in registers; the 4 A fragments per M-tile are loaded once and reused by
    // Q, K and V accumulators (12 independent WMMAs per LDS load batch).
    {
        const int nt = wave;
        v16bf bq[4], bk[4], bv[4];
#pragma unroll
        for (int ks = 0; ks < 4; ++ks) {
            const __bf16* wb = wpack + (size_t)((ks * 8 + nt) * 512) + lane * 16;
            bq[ks] = loadB16(wb);
            bk[ks] = loadB16(wb + 16384);
            bv[ks] = loadB16(wb + 2 * 16384);
        }
#pragma unroll
        for (int mt = 0; mt < 4; ++mt) {
            v16bf a[4];
#pragma unroll
            for (int ks = 0; ks < 4; ++ks)
                a[ks] = loadA(Ts, T_PITCH, mt * 16, ks * 32, lane);
            v8f aq = {}, ak = {}, av = {};
#pragma unroll
            for (int ks = 0; ks < 4; ++ks) {
                aq = wmma_bf16(a[ks], bq[ks], aq);
                ak = wmma_bf16(a[ks], bk[ks], ak);
                av = wmma_bf16(a[ks], bv[ks], av);
            }
            storeD_bf16(Qs, T_PITCH, mt * 16, nt * 16, aq, lane, scale);
            storeD_bf16(Ks, T_PITCH, mt * 16, nt * 16, ak, lane, 1.0f);
            storeD_vt(Vts, mt * 16, nt * 16, av, lane);
        }
    }
    __syncthreads();

    // ---- Phase 3: scores S_h = Q_h * K_h^T (dh=32 -> single WMMA K-step)
    // wave -> (head, key-tile pair); B (K-rows) reused across 4 q-tiles.
    {
        const int h   = wave >> 1;
        const int nt0 = (wave & 1) * 2;
        int key0 = nt0 * 16 + (lane & 15);
        int key1 = key0 + 16;
        v16bf b0 = loadB16(Ks + (size_t)key0 * T_PITCH + h * DH + (lane >> 4) * 16);
        v16bf b1 = loadB16(Ks + (size_t)key1 * T_PITCH + h * DH + (lane >> 4) * 16);
#pragma unroll
        for (int mt = 0; mt < 4; ++mt) {
            v16bf a = loadA(Qs, T_PITCH, mt * 16, h * DH, lane);
            v8f acc0 = {}, acc1 = {};
            acc0 = wmma_bf16(a, b0, acc0);
            acc1 = wmma_bf16(a, b1, acc1);
            storeD_f32(Ss, S_PITCH, h * 64 + mt * 16, nt0 * 16, acc0, lane);
            storeD_f32(Ss, S_PITCH, h * 64 + mt * 16, (nt0 + 1) * 16, acc1, lane);
        }
    }
    __syncthreads();

    // ---- Phase 4: softmax over keys; one thread per (head, q) row; write bf16 P in place
    {
        float* row = Ss + (size_t)tid * S_PITCH;
        float4 v[16];
        float mx = -3.0e38f;
#pragma unroll
        for (int i = 0; i < 16; ++i) {
            v[i] = *(const float4*)(row + i * 4);
            mx = fmaxf(mx, fmaxf(fmaxf(v[i].x, v[i].y), fmaxf(v[i].z, v[i].w)));
        }
        float sum = 0.0f;
#pragma unroll
        for (int i = 0; i < 16; ++i) {
            v[i].x = __expf(v[i].x - mx); v[i].y = __expf(v[i].y - mx);
            v[i].z = __expf(v[i].z - mx); v[i].w = __expf(v[i].w - mx);
            sum += v[i].x + v[i].y + v[i].z + v[i].w;
        }
        float inv = 1.0f / sum;
        __bf16* prow = (__bf16*)row;
#pragma unroll
        for (int i = 0; i < 16; ++i) {
            prow[i * 4 + 0] = (__bf16)(v[i].x * inv);
            prow[i * 4 + 1] = (__bf16)(v[i].y * inv);
            prow[i * 4 + 2] = (__bf16)(v[i].z * inv);
            prow[i * 4 + 3] = (__bf16)(v[i].w * inv);
        }
    }
    __syncthreads();

    // ---- Phase 5: O_h = P_h * V_h (K=64 -> 2 steps); O overlays Q region.
    // wave -> (head, out-col tile); B (Vt) reused across 4 q-tiles; both A fragments
    // per tile are loaded before the WMMA pair.
    {
        const int h  = wave >> 1;
        const int nt = wave & 1;
        int col = h * DH + nt * 16 + (lane & 15);
        v16bf b0 = loadB16(Vts + (size_t)col * VT_PITCH + 0  + (lane >> 4) * 16);
        v16bf b1 = loadB16(Vts + (size_t)col * VT_PITCH + 32 + (lane >> 4) * 16);
#pragma unroll
        for (int mt = 0; mt < 4; ++mt) {
            v16bf a0 = loadA(Ps, T_PITCH, h * 64 + mt * 16, 0,  lane);
            v16bf a1 = loadA(Ps, T_PITCH, h * 64 + mt * 16, 32, lane);
            v8f acc = {};
            acc = wmma_bf16(a0, b0, acc);
            acc = wmma_bf16(a1, b1, acc);
            storeD_bf16(Os, T_PITCH, mt * 16, h * DH + nt * 16, acc, lane, 1.0f);
        }
    }
    __syncthreads();

    // ---- Phase 6: Y = O * Wo, written transposed [d][w]; Y overlays S region.
    // All 4 A fragments per tile loaded before the WMMA chain.
    {
        const int nt = wave;
        v16bf bf[4];
#pragma unroll
        for (int ks = 0; ks < 4; ++ks)
            bf[ks] = loadB16(wpack + 3 * 16384 + (size_t)((ks * 8 + nt) * 512) + lane * 16);
#pragma unroll
        for (int mt = 0; mt < 4; ++mt) {
            v16bf a[4];
#pragma unroll
            for (int ks = 0; ks < 4; ++ks)
                a[ks] = loadA(Os, T_PITCH, mt * 16, ks * 32, lane);
            v8f acc = {};
#pragma unroll
            for (int ks = 0; ks < 4; ++ks)
                acc = wmma_bf16(a[ks], bf[ks], acc);
            storeD_y(Ys, mt * 16, nt * 16, acc, lane);
        }
    }
    __syncthreads();

    // ---- Phase 7: coalesced store of Y[d][w] -> out[b, wn*8192 + d*64 + w]
    {
        int d = tid >> 1, half = tid & 1;
        const float* src = Ys + (size_t)d * S_PITCH + half * 32;
        float* dst = out + (size_t)b * D_MODEL * SEQ_L + (size_t)wn * (D_MODEL * WIN)
                     + (size_t)d * WIN + half * 32;
#pragma unroll
        for (int c = 0; c < 8; ++c)
            *(float4*)(dst + c * 4) = *(const float4*)(src + c * 4);
    }
}

extern "C" void kernel_launch(void* const* d_in, const int* in_sizes, int n_in,
                              void* d_out, int out_size, void* d_ws, size_t ws_size,
                              hipStream_t stream) {
    (void)in_sizes; (void)n_in; (void)out_size; (void)ws_size;
    const float* x  = (const float*)d_in[0];
    const float* Wq = (const float*)d_in[1];
    const float* Wk = (const float*)d_in[2];
    const float* Wv = (const float*)d_in[3];
    const float* Wo = (const float*)d_in[4];
    float* out = (float*)d_out;
    __bf16* wpack = (__bf16*)d_ws;   // 4 * 16384 bf16 = 128 KB

    wa_pack_weights<<<256, 256, 0, stream>>>(Wq, Wk, Wv, Wo, wpack);
    wa_window_kernel<<<8 * NWIN, 256, LDS_BYTES, stream>>>(x, wpack, out);
}
